// GeneralAttention_68092411510947
// MI455X (gfx1250) — compile-verified, matrix operand order checked by reading, not run on hardware
//
#include <hip/hip_runtime.h>

typedef __bf16 bf16_t;
typedef __attribute__((ext_vector_type(4)))  bf16_t v4bf;
typedef __attribute__((ext_vector_type(8)))  bf16_t v8bf;
typedef __attribute__((ext_vector_type(16))) bf16_t v16bf;
typedef __attribute__((ext_vector_type(8)))  float  v8f;

#define LDS_STRIDE 40   // 32 bf16 + 8 pad = 80 B/row, keeps 16B alignment

// ---------------------------------------------------------------------------
// gfx1250 async copy: global -> LDS directly (no VGPR staging), ASYNCcnt.
// ---------------------------------------------------------------------------
__device__ __forceinline__ void async_copy_b128(bf16_t* ldsp, const bf16_t* gp) {
  unsigned loff = (unsigned)(unsigned long long)ldsp;   // LDS offset = low 32 bits
  asm volatile("global_load_async_to_lds_b128 %0, %1, off"
               :: "v"(loff), "v"(gp) : "memory");
}
__device__ __forceinline__ void wait_asynccnt0() {
  asm volatile("s_wait_asynccnt 0x0" ::: "memory");
}

// ---------------------------------------------------------------------------
// Tile loaders: ROWS x 32(K) into padded LDS, 256 threads
// ---------------------------------------------------------------------------

// f32 source (row-major, ld) -> bf16 LDS tile (VGPR staging + cvt)
template <int ROWS>
__device__ __forceinline__ void load_tile_f32_cvt(bf16_t* lds, const float* __restrict__ src,
                                                  int ld, int tid) {
#pragma unroll
  for (int p = 0; p < ROWS / 32; ++p) {
    int e4  = p * 256 + tid;        // float4 index within ROWSx32 tile
    int row = e4 >> 3;              // 8 float4 per row
    int kc  = (e4 & 7) << 2;
    const float4 v = *(const float4*)(src + (size_t)row * ld + kc);
    v4bf b;
    b[0] = (bf16_t)v.x; b[1] = (bf16_t)v.y; b[2] = (bf16_t)v.z; b[3] = (bf16_t)v.w;
    *(v4bf*)(lds + row * LDS_STRIDE + kc) = b;
  }
}

// bf16 source (row-major, ld) -> LDS tile, via async-to-LDS b128 chunks
template <int ROWS>
__device__ __forceinline__ void load_tile_bf16_async(bf16_t* lds, const bf16_t* __restrict__ src,
                                                     int ld, int tid) {
#pragma unroll
  for (int p = 0; p < ROWS / 64; ++p) {
    int e8  = p * 256 + tid;        // 16B chunk index
    int row = e8 >> 2;              // 4 chunks per row
    int kc  = (e8 & 3) << 3;
    async_copy_b128(lds + row * LDS_STRIDE + kc, src + (size_t)row * ld + kc);
  }
}

// f32 source tile 32(K) x NROWS(N) in memory -> transposed lds[n][k] (cvt)
template <int NROWS>
__device__ __forceinline__ void load_tile_f32_T_cvt(bf16_t* lds, const float* __restrict__ src,
                                                    int ld, int tid) {
#pragma unroll
  for (int p = 0; p < NROWS / 32; ++p) {
    int e4 = p * 256 + tid;             // float4 along n (coalesced)
    int k  = e4 / (NROWS / 4);
    int n4 = (e4 % (NROWS / 4)) << 2;
    const float4 v = *(const float4*)(src + (size_t)k * ld + n4);
    lds[(n4 + 0) * LDS_STRIDE + k] = (bf16_t)v.x;
    lds[(n4 + 1) * LDS_STRIDE + k] = (bf16_t)v.y;
    lds[(n4 + 2) * LDS_STRIDE + k] = (bf16_t)v.z;
    lds[(n4 + 3) * LDS_STRIDE + k] = (bf16_t)v.w;
  }
}

// ---------------------------------------------------------------------------
// WMMA fragment load per CDNA5 16-bit A/B layout:
// lanes 0-15: row = lane, K = {0..7, 16..23}; lanes 16-31: K = {8..15, 24..31}
// ---------------------------------------------------------------------------
__device__ __forceinline__ v16bf load_frag16(const bf16_t* lds, int row, int lane) {
  int k0 = (lane >> 4) << 3;
  const bf16_t* p = lds + row * LDS_STRIDE + k0;
  v8bf lo = *(const v8bf*)(p);
  v8bf hi = *(const v8bf*)(p + 16);
  return __builtin_shufflevector(lo, hi, 0, 1, 2, 3, 4, 5, 6, 7,
                                 8, 9, 10, 11, 12, 13, 14, 15);
}

// One BK=32 step: wave (wr, wc) computes a 64x64 patch = 4x4 WMMA tiles
__device__ __forceinline__ void mma_step(const bf16_t* lds_a, const bf16_t* lds_b,
                                         v8f acc[4][4], int lane, int wr, int wc) {
  int lr = lane & 15;
  v16bf af[4], bf_[4];
#pragma unroll
  for (int i = 0; i < 4; ++i) af[i] = load_frag16(lds_a, wr * 64 + i * 16 + lr, lane);
#pragma unroll
  for (int j = 0; j < 4; ++j) bf_[j] = load_frag16(lds_b, wc * 64 + j * 16 + lr, lane);
#pragma unroll
  for (int i = 0; i < 4; ++i)
#pragma unroll
    for (int j = 0; j < 4; ++j)
      acc[i][j] = __builtin_amdgcn_wmma_f32_16x16x32_bf16(
          false, af[i], false, bf_[j], (short)0, acc[i][j], false, false);
}

// Block tile: 128(M) x 256(N) x 32(K); 8 waves in 2x4, 64x64 each.
#define GEMM_PROLOGUE()                                              \
  int tid = threadIdx.x, lane = tid & 31, wave = tid >> 5;           \
  int wr = wave >> 2, wc = wave & 3;                                 \
  v8f acc[4][4] = {};

// ---------------------------------------------------------------------------
// Kernel 1: key_t = bf16(key @ W^T)   M=16384, N=1024, K=1024 (both f32 in)
// ---------------------------------------------------------------------------
__global__ __launch_bounds__(256) void kt_gemm_kernel(const float* __restrict__ A,
                                                      const float* __restrict__ Wm,
                                                      bf16_t* __restrict__ out) {
  constexpr int K = 1024, N = 1024;
  __shared__ bf16_t sA[2][128 * LDS_STRIDE];
  __shared__ bf16_t sB[2][256 * LDS_STRIDE];
  GEMM_PROLOGUE();
  const float* aT = A  + (size_t)blockIdx.y * 128 * K;
  const float* bT = Wm + (size_t)blockIdx.x * 256 * K;

  load_tile_f32_cvt<128>(sA[0], aT, K, tid);
  load_tile_f32_cvt<256>(sB[0], bT, K, tid);
  __syncthreads();
  for (int kb = 0; kb < K; kb += 32) {
    int cur = (kb >> 5) & 1;
    if (kb + 32 < K) {
      load_tile_f32_cvt<128>(sA[cur ^ 1], aT + kb + 32, K, tid);
      load_tile_f32_cvt<256>(sB[cur ^ 1], bT + kb + 32, K, tid);
    }
    mma_step(sA[cur], sB[cur], acc, lane, wr, wc);
    __syncthreads();
  }

  int rbase = blockIdx.y * 128 + wr * 64 + ((lane >> 4) << 3);
  int cbase = blockIdx.x * 256 + wc * 64 + (lane & 15);
#pragma unroll
  for (int i = 0; i < 4; ++i)
#pragma unroll
    for (int j = 0; j < 4; ++j)
#pragma unroll
      for (int r = 0; r < 8; ++r)
        out[(size_t)(rbase + i * 16 + r) * N + cbase + j * 16] = (bf16_t)acc[i][j][r];
}

// ---------------------------------------------------------------------------
// Kernel 2: energy = (Q @ key_t^T)/32, masked   per batch M=2048,N=2048,K=1024
// A: f32 cvt path; B: bf16 async-to-LDS path
// ---------------------------------------------------------------------------
__global__ __launch_bounds__(256) void qk_gemm_kernel(const float* __restrict__ Q,
                                                      const bf16_t* __restrict__ Kt,
                                                      const int* __restrict__ mask,
                                                      float* __restrict__ att) {
  constexpr int K = 1024, Lk = 2048;
  __shared__ bf16_t sA[2][128 * LDS_STRIDE];
  __shared__ bf16_t sB[2][256 * LDS_STRIDE];
  GEMM_PROLOGUE();
  int b = blockIdx.z;
  const float*  aT = Q  + (size_t)b * 2048 * K + (size_t)blockIdx.y * 128 * K;
  const bf16_t* bT = Kt + (size_t)b * 2048 * K + (size_t)blockIdx.x * 256 * K;

  load_tile_f32_cvt<128>(sA[0], aT, K, tid);
  load_tile_bf16_async<256>(sB[0], bT, K, tid);
  wait_asynccnt0();
  __syncthreads();
  for (int kb = 0; kb < K; kb += 32) {
    int cur = (kb >> 5) & 1;
    if (kb + 32 < K) {
      load_tile_f32_cvt<128>(sA[cur ^ 1], aT + kb + 32, K, tid);
      load_tile_bf16_async<256>(sB[cur ^ 1], bT + kb + 32, K, tid);
    }
    mma_step(sA[cur], sB[cur], acc, lane, wr, wc);   // overlaps async copy
    wait_asynccnt0();
    __syncthreads();
  }

  const float scale = 0.03125f;   // 1/sqrt(1024)
  const float NEGV  = -1e10f;
  int rbase = blockIdx.y * 128 + wr * 64 + ((lane >> 4) << 3);
  int cbase = blockIdx.x * 256 + wc * 64 + (lane & 15);
  size_t boff = (size_t)b * 2048 * Lk;
#pragma unroll
  for (int i = 0; i < 4; ++i)
#pragma unroll
    for (int j = 0; j < 4; ++j)
#pragma unroll
      for (int r = 0; r < 8; ++r) {
        size_t off = boff + (size_t)(rbase + i * 16 + r) * Lk + cbase + j * 16;
        float e = acc[i][j][r] * scale;
        att[off] = (mask[off] != 0) ? e : NEGV;
      }
}

// ---------------------------------------------------------------------------
// Kernel 3: row softmax over Lk=2048 in place (f32) + bf16 copy for A@V
// ---------------------------------------------------------------------------
__global__ __launch_bounds__(256) void softmax_kernel(float* __restrict__ att,
                                                      bf16_t* __restrict__ att_bf) {
  constexpr int Lk = 2048;
  __shared__ float red[256];
  int tid = threadIdx.x;
  size_t roff = (size_t)blockIdx.x * Lk;
  float* p = att + roff;
  float vals[8];
  float m = -3.402823466e38f;
#pragma unroll
  for (int i = 0; i < 8; ++i) { vals[i] = p[tid + i * 256]; m = fmaxf(m, vals[i]); }
  red[tid] = m;
  __syncthreads();
  for (int s = 128; s > 0; s >>= 1) {
    if (tid < s) red[tid] = fmaxf(red[tid], red[tid + s]);
    __syncthreads();
  }
  float rowmax = red[0];
  __syncthreads();
  float sum = 0.f;
#pragma unroll
  for (int i = 0; i < 8; ++i) { vals[i] = __expf(vals[i] - rowmax); sum += vals[i]; }
  red[tid] = sum;
  __syncthreads();
  for (int s = 128; s > 0; s >>= 1) {
    if (tid < s) red[tid] += red[tid + s];
    __syncthreads();
  }
  float inv = 1.f / red[0];
#pragma unroll
  for (int i = 0; i < 8; ++i) {
    float a = vals[i] * inv;
    p[tid + i * 256] = a;
    att_bf[roff + tid + i * 256] = (bf16_t)a;
  }
}

// ---------------------------------------------------------------------------
// Kernel 4: context = att_bf16 @ V   per batch M=2048, N=1024, K=2048
// A: bf16 async-to-LDS path; B: f32 transposed cvt path
// ---------------------------------------------------------------------------
__global__ __launch_bounds__(256) void av_gemm_kernel(const bf16_t* __restrict__ Ab,
                                                      const float* __restrict__ V,
                                                      float* __restrict__ ctx) {
  constexpr int K = 2048, N = 1024;
  __shared__ bf16_t sA[2][128 * LDS_STRIDE];
  __shared__ bf16_t sB[2][256 * LDS_STRIDE];
  GEMM_PROLOGUE();
  int b = blockIdx.z;
  const bf16_t* aT = Ab + (size_t)b * 2048 * K + (size_t)blockIdx.y * 128 * K;
  const float*  vB = V  + (size_t)b * 2048 * N + blockIdx.x * 256;

  load_tile_bf16_async<128>(sA[0], aT, K, tid);
  load_tile_f32_T_cvt<256>(sB[0], vB, N, tid);
  wait_asynccnt0();
  __syncthreads();
  for (int kb = 0; kb < K; kb += 32) {
    int cur = (kb >> 5) & 1;
    if (kb + 32 < K) {
      load_tile_bf16_async<128>(sA[cur ^ 1], aT + kb + 32, K, tid);
      load_tile_f32_T_cvt<256>(sB[cur ^ 1], vB + (size_t)(kb + 32) * N, N, tid);
    }
    mma_step(sA[cur], sB[cur], acc, lane, wr, wc);   // overlaps async copy
    wait_asynccnt0();
    __syncthreads();
  }

  int rbase = blockIdx.y * 128 + wr * 64 + ((lane >> 4) << 3);
  int cbase = blockIdx.x * 256 + wc * 64 + (lane & 15);
  size_t boff = (size_t)b * 2048 * N;
#pragma unroll
  for (int i = 0; i < 4; ++i)
#pragma unroll
    for (int j = 0; j < 4; ++j)
#pragma unroll
      for (int r = 0; r < 8; ++r)
        ctx[boff + (size_t)(rbase + i * 16 + r) * N + cbase + j * 16] = acc[i][j][r];
}

// ---------------------------------------------------------------------------
extern "C" void kernel_launch(void* const* d_in, const int* in_sizes, int n_in,
                              void* d_out, int out_size, void* d_ws, size_t ws_size,
                              hipStream_t stream) {
  (void)in_sizes; (void)n_in; (void)out_size; (void)ws_size;
  const float* query = (const float*)d_in[0];   // [8,2048,1024]
  const float* key   = (const float*)d_in[1];   // [8,2048,1024]
  const float* value = (const float*)d_in[2];   // [8,2048,1024]
  const int*   mask  = (const int*)  d_in[3];   // [8,2048,2048]
  const float* Wm    = (const float*)d_in[4];   // [1024,1024]

  float* context = (float*)d_out;                        // 8*2048*1024 f32
  float* att     = context + (size_t)8 * 2048 * 1024;    // 8*2048*2048 f32

  bf16_t* key_t  = (bf16_t*)d_ws;                        // 33.5 MB bf16
  bf16_t* att_bf = key_t + (size_t)8 * 2048 * 1024;      // 67 MB bf16

  kt_gemm_kernel<<<dim3(1024 / 256, (8 * 2048) / 128, 1), 256, 0, stream>>>(key, Wm, key_t);
  qk_gemm_kernel<<<dim3(2048 / 256, 2048 / 128, 8), 256, 0, stream>>>(query, key_t, mask, att);
  softmax_kernel<<<dim3(8 * 2048), 256, 0, stream>>>(att, att_bf);
  av_gemm_kernel<<<dim3(1024 / 256, 2048 / 128, 8), 256, 0, stream>>>(att_bf, value, context);
}
